// AttentionMLP_11656541241543
// MI455X (gfx1250) — matrix operand order repaired and checked by reading, not compiled, not run on hardware
//
#include <hip/hip_runtime.h>
#include <hip/hip_bf16.h>

typedef __attribute__((ext_vector_type(16))) __bf16 v16bf;
typedef __attribute__((ext_vector_type(8)))  __bf16 v8bf;
typedef __attribute__((ext_vector_type(8)))  float  v8f;

#define B_     32
#define C_     640
#define N_     1024
#define HEADS_ 16
#define DH_    64
#define INNER_ 1024

// ---------------------------------------------------------------------------
// 1) x [B,C,N] fp32  ->  t [B,N,C] bf16   (LDS-tiled transpose + convert)
// ---------------------------------------------------------------------------
__global__ __launch_bounds__(256)
void xpose_bf16(const float* __restrict__ x, __bf16* __restrict__ t) {
  __shared__ float tile[32][33];
  const int b  = blockIdx.z;
  const int c0 = blockIdx.x * 32;   // C/32 = 20
  const int n0 = blockIdx.y * 32;   // N/32 = 32
  for (int i = threadIdx.y; i < 32; i += 8)
    tile[i][threadIdx.x] = x[((size_t)b * C_ + (c0 + i)) * N_ + n0 + threadIdx.x];
  __syncthreads();
  for (int i = threadIdx.y; i < 32; i += 8)
    t[((size_t)b * N_ + (n0 + i)) * C_ + c0 + threadIdx.x] =
        (__bf16)tile[threadIdx.x][i];
}

// ---------------------------------------------------------------------------
// 1b) Wq/Wk/Wv fp32 [1024,640] -> one bf16 [3072,640]
// ---------------------------------------------------------------------------
__global__ void wconv(const float* __restrict__ Wq, const float* __restrict__ Wk,
                      const float* __restrict__ Wv, __bf16* __restrict__ Wall) {
  size_t idx = (size_t)blockIdx.x * 256 + threadIdx.x;
  const size_t per = (size_t)INNER_ * C_;
  const float* src = (idx < per) ? Wq : (idx < 2 * per ? Wk : Wv);
  Wall[idx] = (__bf16)src[idx % per];
}

// ---------------------------------------------------------------------------
// 2) Projection GEMM: [32768 x 640] x [640 x 3072], bf16 WMMA, fp32 accum.
//    Block = 256 thr (8 waves), 128x64 tile; wave = 16x64 strip:
//    one A-frag reused across 4 B-frags -> 4 WMMAs / wave / K-step.
// ---------------------------------------------------------------------------
__global__ __launch_bounds__(256)
void proj_wmma(const __bf16* __restrict__ T, const __bf16* __restrict__ Wall,
               __bf16* __restrict__ QKV) {
  __shared__ __attribute__((aligned(16))) __bf16 As[128 * 40]; // [m][k] pad 40
  __shared__ __attribute__((aligned(16))) __bf16 Bs[64 * 40];  // [n][k] pad 40
  const int tid  = threadIdx.x;
  const int lane = tid & 31;
  const int w    = tid >> 5;
  const int hi   = lane >> 4;
  const int ln   = lane & 15;
  const int row0 = blockIdx.x * 128;
  const int col0 = blockIdx.y * 64;

  v8f acc[4] = {};
  for (int kk = 0; kk < C_; kk += 32) {
    __syncthreads();
    // A tile 128x32 = 512 x 16B chunks, 2 per thread
#pragma unroll
    for (int i = 0; i < 2; ++i) {
      int c = i * 256 + tid;
      int m = c >> 2, kc = c & 3;
      *(v8bf*)&As[m * 40 + kc * 8] =
          *(const v8bf*)&T[(size_t)(row0 + m) * C_ + kk + kc * 8];
    }
    // B tile 64x32 = 256 x 16B chunks, 1 per thread
    {
      int m = tid >> 2, kc = tid & 3;
      *(v8bf*)&Bs[m * 40 + kc * 8] =
          *(const v8bf*)&Wall[(size_t)(col0 + m) * C_ + kk + kc * 8];
    }
    __syncthreads();

    v16bf a;
    const __bf16* ap = &As[(w * 16 + ln) * 40];
#pragma unroll
    for (int e = 0; e < 16; ++e)
      a[e] = ap[16 * hi + e];         // A-frag layout
#pragma unroll
    for (int nt = 0; nt < 4; ++nt) {
      v16bf bfr;
      const __bf16* bp = &Bs[(nt * 16 + ln) * 40 + 16 * hi];
#pragma unroll
      for (int e = 0; e < 16; ++e) bfr[e] = bp[e];          // B-frag layout
      acc[nt] = __builtin_amdgcn_wmma_f32_16x16x32_bf16(
          false, a, false, bfr, (short)0, acc[nt], false, false);
    }
  }
#pragma unroll
  for (int nt = 0; nt < 4; ++nt) {
#pragma unroll
    for (int r = 0; r < 8; ++r) {          // C/D layout scatter -> bf16
      int grow = row0 + w * 16 + r + 8 * hi;
      int gcol = col0 + nt * 16 + ln;
      int b = grow >> 10, npos = grow & 1023;
      int p = gcol >> 10, e = gcol & 1023;
      int h = e >> 6, dd = e & 63;
      QKV[((((size_t)p * B_ + b) * HEADS_ + h) * N_ + npos) * DH_ + dd] =
          (__bf16)acc[nt][r];
    }
  }
}

// ---------------------------------------------------------------------------
// 3) Flash attention + query-mean accumulation.
//    Block = 256 thr (8 waves), one (b,h) slab of 128 queries; wave = 16 rows.
// ---------------------------------------------------------------------------
__global__ __launch_bounds__(256)
void attn_wmma(const __bf16* __restrict__ QKV, float* __restrict__ osum) {
  __shared__ __attribute__((aligned(16))) __bf16 Ks[32 * 72];   // [key][d]
  __shared__ __attribute__((aligned(16))) __bf16 Vs[64 * 40];   // [d][key]
  __shared__ __attribute__((aligned(16))) __bf16 Ps[8][16 * 40]; // per-wave P
  const int tid  = threadIdx.x;
  const int lane = tid & 31;
  const int w    = tid >> 5;
  const int hi   = lane >> 4;
  const int ln   = lane & 15;
  const int bh   = blockIdx.x;                 // b*16 + h
  const int q0   = blockIdx.y * 128 + w * 16;
  const size_t plane = (size_t)B_ * HEADS_ * N_ * DH_;
  const __bf16* Qg = QKV + (size_t)bh * N_ * DH_;
  const __bf16* Kg = QKV + plane + (size_t)bh * N_ * DH_;
  const __bf16* Vg = QKV + 2 * plane + (size_t)bh * N_ * DH_;

  // Q fragments: per lane two contiguous 16B runs per K=32 fragment
  v16bf aq[2];
  const __bf16* qp = Qg + (size_t)(q0 + ln) * DH_;
#pragma unroll
  for (int kk = 0; kk < 2; ++kk) {
    v8bf lo  = *(const v8bf*)&qp[kk * 32 + 16 * hi];
    v8bf hi8 = *(const v8bf*)&qp[kk * 32 + 16 * hi + 8];
#pragma unroll
    for (int e = 0; e < 8; ++e) { aq[kk][e] = lo[e]; aq[kk][8 + e] = hi8[e]; }
  }

  float mrow[8], lrow[8];
  v8f o[4] = {};
#pragma unroll
  for (int r = 0; r < 8; ++r) { mrow[r] = -1e30f; lrow[r] = 0.f; }
  const float scale = 0.125f;  // 64^-0.5

  for (int kb = 0; kb < N_; kb += 32) {
    __syncthreads();
    {   // K tile 32x64: one 16B chunk per thread
      int key = tid >> 3, kc = tid & 7;
      *(v8bf*)&Ks[key * 72 + kc * 8] =
          *(const v8bf*)&Kg[(size_t)(kb + key) * DH_ + kc * 8];
      // V tile: vector load, transpose-scatter into [d][key]
      v8bf vv = *(const v8bf*)&Vg[(size_t)(kb + key) * DH_ + kc * 8];
#pragma unroll
      for (int j = 0; j < 8; ++j) Vs[(kc * 8 + j) * 40 + key] = vv[j];
    }
    if (kb + 32 < N_) {               // hint next tiles toward the caches
      __builtin_prefetch(Kg + (size_t)(kb + 32) * DH_ + tid * 8, 0, 3);
      __builtin_prefetch(Vg + (size_t)(kb + 32) * DH_ + tid * 8, 0, 3);
    }
    __syncthreads();

    // S[16x32] = Q . K^T   (two 16-col halves, K-dim = 64 in two steps)
    v8f s0v = {}, s1v = {};
#pragma unroll
    for (int kk = 0; kk < 2; ++kk) {
      v16bf b0, b1;
      const __bf16* bp0 = &Ks[ln * 72 + kk * 32 + 16 * hi];
      const __bf16* bp1 = &Ks[(16 + ln) * 72 + kk * 32 + 16 * hi];
#pragma unroll
      for (int e = 0; e < 16; ++e) { b0[e] = bp0[e]; b1[e] = bp1[e]; }
      s0v = __builtin_amdgcn_wmma_f32_16x16x32_bf16(false, aq[kk], false, b0,
                                                    (short)0, s0v, false, false);
      s1v = __builtin_amdgcn_wmma_f32_16x16x32_bf16(false, aq[kk], false, b1,
                                                    (short)0, s1v, false, false);
    }

    // online softmax (row reductions across the 16-lane half via shfl_xor)
    float p0[8], p1[8];
#pragma unroll
    for (int r = 0; r < 8; ++r) {
      float a0 = s0v[r] * scale, a1 = s1v[r] * scale;
      float tmax = fmaxf(a0, a1);
#pragma unroll
      for (int msk = 1; msk <= 8; msk <<= 1)
        tmax = fmaxf(tmax, __shfl_xor(tmax, msk, 32));
      float mnew  = fmaxf(mrow[r], tmax);
      float alpha = __expf(mrow[r] - mnew);
      mrow[r] = mnew;
      p0[r] = __expf(a0 - mnew);
      p1[r] = __expf(a1 - mnew);
      float rs = p0[r] + p1[r];
#pragma unroll
      for (int msk = 1; msk <= 8; msk <<= 1)
        rs += __shfl_xor(rs, msk, 32);
      lrow[r] = lrow[r] * alpha + rs;
#pragma unroll
      for (int nn = 0; nn < 4; ++nn) o[nn][r] *= alpha;
    }

    // P -> wave-private LDS (C-layout scatter), re-read as A-fragment
    __bf16* pw = &Ps[w][0];
#pragma unroll
    for (int r = 0; r < 8; ++r) {
      int row = r + 8 * hi;
      pw[row * 40 + ln]      = (__bf16)p0[r];
      pw[row * 40 + 16 + ln] = (__bf16)p1[r];
    }
    v16bf pa;
    const __bf16* pr = &Ps[w][ln * 40];
#pragma unroll
    for (int e = 0; e < 16; ++e)
      pa[e] = pr[16 * hi + e];

    // O[16x64] += P . V   (4 column tiles of 16)
#pragma unroll
    for (int nn = 0; nn < 4; ++nn) {
      v16bf bv;
      const __bf16* vp = &Vs[(nn * 16 + ln) * 40 + 16 * hi];
#pragma unroll
      for (int e = 0; e < 16; ++e) bv[e] = vp[e];
      o[nn] = __builtin_amdgcn_wmma_f32_16x16x32_bf16(false, pa, false, bv,
                                                      (short)0, o[nn], false, false);
    }
  }

  // sum over this wave's 16 query rows (divide by row softmax sum), accumulate
#pragma unroll
  for (int nn = 0; nn < 4; ++nn) {
    float cs = 0.f;
#pragma unroll
    for (int r = 0; r < 8; ++r) cs += o[nn][r] / lrow[r];
    cs += __shfl_xor(cs, 16, 32);        // fold the other 8 rows
    if (lane < 16)
      atomicAdd(&osum[(size_t)bh * DH_ + nn * 16 + ln], cs);
  }
}

// ---------------------------------------------------------------------------
// 4) mean over N + LayerNorm(64) -> out [B, heads, 64] fp32
// ---------------------------------------------------------------------------
__global__ __launch_bounds__(64)
void finalize_ln(const float* __restrict__ osum, const float* __restrict__ gamma,
                 const float* __restrict__ beta, float* __restrict__ out) {
  __shared__ float red[64];
  const int row = blockIdx.x, t = threadIdx.x;
  float v = osum[row * 64 + t] * (1.0f / 1024.0f);
  red[t] = v;
  __syncthreads();
  for (int s = 32; s > 0; s >>= 1) { if (t < s) red[t] += red[t + s]; __syncthreads(); }
  float mu = red[0] * (1.0f / 64.0f);
  __syncthreads();
  float d = v - mu;
  red[t] = d * d;
  __syncthreads();
  for (int s = 32; s > 0; s >>= 1) { if (t < s) red[t] += red[t + s]; __syncthreads(); }
  float var = red[0] * (1.0f / 64.0f);
  out[row * 64 + t] = d * rsqrtf(var + 1e-5f) * gamma[t] + beta[t];
}

// ---------------------------------------------------------------------------
extern "C" void kernel_launch(void* const* d_in, const int* in_sizes, int n_in,
                              void* d_out, int out_size, void* d_ws, size_t ws_size,
                              hipStream_t stream) {
  const float* x     = (const float*)d_in[0];
  const float* Wq    = (const float*)d_in[1];
  const float* Wk    = (const float*)d_in[2];
  const float* Wv    = (const float*)d_in[3];
  const float* gamma = (const float*)d_in[4];
  const float* beta  = (const float*)d_in[5];
  float* out = (float*)d_out;

  char* ws = (char*)d_ws;
  size_t off = 0;
  auto carve = [&](size_t bytes) -> void* {
    void* p = ws + off;
    off = (off + bytes + 255) & ~(size_t)255;
    return p;
  };
  __bf16* Tb   = (__bf16*)carve((size_t)B_ * N_ * C_ * 2);              // 40 MB
  __bf16* Wb   = (__bf16*)carve((size_t)3 * INNER_ * C_ * 2);           // 3.75 MB
  __bf16* QKV  = (__bf16*)carve((size_t)3 * B_ * HEADS_ * N_ * DH_ * 2);// 192 MB
  float*  osum = (float*)carve((size_t)B_ * HEADS_ * DH_ * 4);          // 128 KB

  hipMemsetAsync(osum, 0, (size_t)B_ * HEADS_ * DH_ * 4, stream);

  xpose_bf16<<<dim3(C_ / 32, N_ / 32, B_), dim3(32, 8), 0, stream>>>(x, Tb);
  wconv<<<(3 * INNER_ * C_) / 256, 256, 0, stream>>>(Wq, Wk, Wv, Wb);
  proj_wmma<<<dim3((B_ * N_) / 128, (3 * INNER_) / 64), 256, 0, stream>>>(Tb, Wb, QKV);
  attn_wmma<<<dim3(B_ * HEADS_, N_ / 128), 256, 0, stream>>>(QKV, osum);
  finalize_ln<<<B_ * HEADS_, 64, 0, stream>>>(osum, gamma, beta, out);
}
